// LiftSplatShoot_75179107549309
// MI455X (gfx1250) — compile-verified, matrix operand order checked by reading, not run on hardware
//
#include <hip/hip_runtime.h>
#include <hip/hip_bf16.h>

typedef __attribute__((ext_vector_type(16))) __bf16         v16bf;
typedef __attribute__((ext_vector_type(8)))  float          v8f;
typedef __attribute__((ext_vector_type(8)))  unsigned short us8;
typedef __attribute__((ext_vector_type(16))) unsigned short us16;

#define HW      11264   // 64 * 176
#define IMG_W   176
#define IMG_H   64
#define C_IN    2048
#define C_MID   512
#define K1      18432   // 2048 * 9
#define DB      64
// GEMM1 tiles
#define BM1     64
#define BN1     128
#define LDAT    40      // 32 k-slots + 8 pad (ushorts); row = 80B (16B-aligned)
#define LDBT    40

static __device__ __forceinline__ unsigned short f32_to_bf16(float f) {
    unsigned int u = __float_as_uint(f);
    u += 0x7FFFu + ((u >> 16) & 1u);   // round-to-nearest-even
    return (unsigned short)(u >> 16);
}
// WMMA 16-bit lane order permutation: swap bits 3 and 4 (involution).
// Storage slot (lhi*16 + j) then holds k = j + (j&8) + 8*lhi as required.
static __device__ __forceinline__ int posk(int k) {
    return (k & 7) | ((k & 8) << 1) | ((k & 16) >> 1);
}
static __device__ __forceinline__ v16bf load_frag(const unsigned short* p) {
    us8 lo = *(const us8*)(p);       // ds_load_b128
    us8 hi = *(const us8*)(p + 8);   // ds_load_b128
    us16 v = __builtin_shufflevector(lo, hi, 0,1,2,3,4,5,6,7,8,9,10,11,12,13,14,15);
    return __builtin_bit_cast(v16bf, v);
}

// ---------------- utility kernels ----------------
__global__ void zero_f32v(float4* p, size_t n4) {
    size_t i = (size_t)blockIdx.x * blockDim.x + threadIdx.x;
    size_t s = (size_t)gridDim.x * blockDim.x;
    float4 z = {0.f, 0.f, 0.f, 0.f};
    for (; i < n4; i += s) p[i] = z;
}

// feat [2048][11264] f32 -> featP [11264][2048] bf16, channel index permuted per 32-block
__global__ __launch_bounds__(256) void cvt_feat_t(const float* __restrict__ src,
                                                  unsigned short* __restrict__ dst) {
    __shared__ float tile[32][33];
    const int p0 = blockIdx.x * 32;      // pixel tile
    const int c0 = blockIdx.y * 32;      // channel tile
    const int tx = threadIdx.x & 31;
    const int ty = threadIdx.x >> 5;     // 0..7
#pragma unroll
    for (int i = 0; i < 32; i += 8)
        tile[ty + i][tx] = src[(size_t)(c0 + ty + i) * HW + p0 + tx];
    __syncthreads();
    const int cp = posk(tx);             // slot tx holds channel c0+posk(tx)
#pragma unroll
    for (int i = 0; i < 32; i += 8)
        dst[(size_t)(p0 + ty + i) * C_IN + c0 + tx] = f32_to_bf16(tile[cp][ty + i]);
}

// w1 (512,2048,3,3) OIHW f32 -> wBP [n][K1] bf16, k = tap*2048+c, permuted per 32-block
__global__ void cvt_w1_bf16(const float* __restrict__ w1, unsigned short* __restrict__ dst) {
    size_t i = (size_t)blockIdx.x * blockDim.x + threadIdx.x;
    size_t s = (size_t)gridDim.x * blockDim.x;
    const size_t n_el = (size_t)C_MID * K1;
    for (; i < n_el; i += s) {
        const int n = (int)(i / K1);
        const int r = (int)(i - (size_t)n * K1);
        const int k = (r & ~31) | posk(r & 31);
        const int tap = k >> 11;
        const int c   = k & 2047;
        dst[i] = f32_to_bf16(w1[(size_t)n * K1 + (size_t)c * 9 + tap]);
    }
}

// w2 (64,512,1,1) f32 -> w2bP [n][512] bf16, k permuted per 32-block
__global__ void cvt_w2_bf16(const float* __restrict__ w2, unsigned short* __restrict__ dst) {
    int i = blockIdx.x * blockDim.x + threadIdx.x;
    if (i < DB * C_MID) {
        const int n = i >> 9;
        const int r = i & 511;
        const int k = (r & ~31) | posk(r & 31);
        dst[i] = f32_to_bf16(w2[(size_t)n * C_MID + k]);
    }
}

// ---------------- GEMM1: fused im2col conv3x3 + bias + BN + ReLU ----------------
// M=11264, N=512, K=18432. Block 64x128, 8 waves, each wave 32x32 (4 wmma/chunk).
__global__ __launch_bounds__(256) void gemm1_conv3x3(
    const unsigned short* __restrict__ featP,    // [11264][2048] bf16 (permuted)
    const unsigned short* __restrict__ wBP,      // [512][18432]  bf16 (permuted)
    const unsigned short* __restrict__ zrow,     // 2048 zeros (border guard)
    const float* __restrict__ b1,
    const float* __restrict__ gamma, const float* __restrict__ beta,
    const float* __restrict__ mean,  const float* __restrict__ var,
    unsigned short* __restrict__ a2)             // [11264][512] bf16 (permuted cols)
{
    __shared__ __align__(16) unsigned short lds_aT[BM1 * LDAT];  // [m][slot]
    __shared__ __align__(16) unsigned short lds_bT[BN1 * LDBT];  // [n][slot]

    const int tid  = threadIdx.x;
    const int lane = tid & 31;
    const int wv   = tid >> 5;
    const int lrow = lane & 15;
    const int lhi  = lane >> 4;
    const int m0   = blockIdx.x * BM1;
    const int n0   = blockIdx.y * BN1;
    const int wm   = wv >> 2;          // 0..1 -> 32 rows
    const int wn   = wv & 3;           // 0..3 -> 32 cols

    // A staging: one us8 per thread: row a_ml, slots a_s8..a_s8+7
    const int a_ml = tid >> 2;
    const int a_s8 = (tid & 3) * 8;
    const int am   = m0 + a_ml;
    const int apy  = am / IMG_W;
    const int apx  = am % IMG_W;
    // B staging: two us8 per thread: rows b_n and b_n+64
    const int b_n  = tid >> 2;
    const int b_s8 = (tid & 3) * 8;

    unsigned short* stA  = &lds_aT[a_ml * LDAT + a_s8];
    unsigned short* stB0 = &lds_bT[b_n * LDBT + b_s8];
    unsigned short* stB1 = &lds_bT[(b_n + 64) * LDBT + b_s8];

    v8f acc00 = {}, acc01 = {}, acc10 = {}, acc11 = {};

    for (int tap = 0; tap < 9; ++tap) {
        const int dy = tap / 3 - 1;
        const int dx = tap % 3 - 1;
        const int py = apy + dy;
        const int px = apx + dx;
        const bool avalid = ((unsigned)py < (unsigned)IMG_H) & ((unsigned)px < (unsigned)IMG_W);
        const unsigned short* rowA =
            avalid ? (featP + (size_t)(py * IMG_W + px) * C_IN + a_s8) : (zrow + a_s8);
        const unsigned short* rowB0 = wBP + (size_t)(n0 + b_n) * K1 + tap * 2048 + b_s8;
        const unsigned short* rowB1 = rowB0 + (size_t)64 * K1;

        for (int cb = 0; cb < 64; ++cb) {
            const int c0 = cb * 32;
            us8 va  = *(const us8*)(rowA  + c0);   // global_load_b128
            us8 vb0 = *(const us8*)(rowB0 + c0);
            us8 vb1 = *(const us8*)(rowB1 + c0);
            *(us8*)stA  = va;                       // ds_store_b128
            *(us8*)stB0 = vb0;
            *(us8*)stB1 = vb1;
            __syncthreads();

            if (cb + 1 < 64)
                __builtin_prefetch(rowB0 + c0 + 32, 0, 1);   // global_prefetch

            // fragments: contiguous 32B per lane (2x ds_load_b128 each)
            v16bf a0  = load_frag(&lds_aT[(wm * 32 +  0 + lrow) * LDAT + lhi * 16]);
            v16bf a1  = load_frag(&lds_aT[(wm * 32 + 16 + lrow) * LDAT + lhi * 16]);
            v16bf bf0 = load_frag(&lds_bT[(wn * 32 +  0 + lrow) * LDBT + lhi * 16]);
            v16bf bf1 = load_frag(&lds_bT[(wn * 32 + 16 + lrow) * LDBT + lhi * 16]);

            acc00 = __builtin_amdgcn_wmma_f32_16x16x32_bf16(false, a0, false, bf0, (short)0, acc00, false, false);
            acc01 = __builtin_amdgcn_wmma_f32_16x16x32_bf16(false, a0, false, bf1, (short)0, acc01, false, false);
            acc10 = __builtin_amdgcn_wmma_f32_16x16x32_bf16(false, a1, false, bf0, (short)0, acc10, false, false);
            acc11 = __builtin_amdgcn_wmma_f32_16x16x32_bf16(false, a1, false, bf1, (short)0, acc11, false, false);
            __syncthreads();
        }
    }

    // fused epilogue: bias + BN + ReLU; store bf16 with permuted column slot
#pragma unroll
    for (int i = 0; i < 2; ++i) {
#pragma unroll
        for (int j = 0; j < 2; ++j) {
            const v8f acc = (i == 0) ? (j == 0 ? acc00 : acc01) : (j == 0 ? acc10 : acc11);
            const int nloc = j * 16 + lrow;                    // 0..31 within wave cols
            const int gn   = n0 + wn * 32 + nloc;              // true channel
            const int pcol = n0 + wn * 32 + posk(nloc);        // storage slot
            const float sc = gamma[gn] * rsqrtf(var[gn] + 1e-5f);
            const float sh = beta[gn] - sc * mean[gn] + sc * b1[gn];
#pragma unroll
            for (int r = 0; r < 8; ++r) {
                const int gm = m0 + wm * 32 + i * 16 + r + lhi * 8;
                float v = sc * acc[r] + sh;
                v = v > 0.0f ? v : 0.0f;
                a2[(size_t)gm * C_MID + pcol] = f32_to_bf16(v);
            }
        }
    }
}

// ---------------- GEMM2: conv1x1 (512 -> 64) + bias ----------------
// M=11264, N=64, K=512. Block 64x64, 8 waves (2Mx4N), each wave 32x16 (2 wmma/chunk).
__global__ __launch_bounds__(256) void gemm2_conv1x1(
    const unsigned short* __restrict__ a2,    // [11264][512] bf16 (permuted cols)
    const unsigned short* __restrict__ w2bP,  // [64][512]    bf16 (permuted)
    const float* __restrict__ b2,
    float* __restrict__ logits)               // [11264][64]  f32
{
    __shared__ __align__(16) unsigned short lds_aT[64 * LDAT];
    __shared__ __align__(16) unsigned short lds_bT[64 * LDBT];

    const int tid  = threadIdx.x;
    const int lane = tid & 31;
    const int wv   = tid >> 5;
    const int lrow = lane & 15;
    const int lhi  = lane >> 4;
    const int m0   = blockIdx.x * 64;
    const int wm   = wv >> 2;     // 0..1
    const int wn   = wv & 3;      // 0..3 -> 16 cols each

    const int s_r  = tid >> 2;            // row (m or n)
    const int s_s8 = (tid & 3) * 8;       // slot start

    const unsigned short* rowA = a2   + (size_t)(m0 + s_r) * C_MID + s_s8;
    const unsigned short* rowB = w2bP + (size_t)s_r * C_MID + s_s8;
    unsigned short* stA = &lds_aT[s_r * LDAT + s_s8];
    unsigned short* stB = &lds_bT[s_r * LDBT + s_s8];

    v8f acc0 = {}, acc1 = {};

    for (int cb = 0; cb < 16; ++cb) {
        const int k0 = cb * 32;
        us8 va = *(const us8*)(rowA + k0);
        us8 vb = *(const us8*)(rowB + k0);
        *(us8*)stA = va;
        *(us8*)stB = vb;
        __syncthreads();

        v16bf a0  = load_frag(&lds_aT[(wm * 32 +  0 + lrow) * LDAT + lhi * 16]);
        v16bf a1  = load_frag(&lds_aT[(wm * 32 + 16 + lrow) * LDAT + lhi * 16]);
        v16bf bf0 = load_frag(&lds_bT[(wn * 16 + lrow) * LDBT + lhi * 16]);

        acc0 = __builtin_amdgcn_wmma_f32_16x16x32_bf16(false, a0, false, bf0, (short)0, acc0, false, false);
        acc1 = __builtin_amdgcn_wmma_f32_16x16x32_bf16(false, a1, false, bf0, (short)0, acc1, false, false);
        __syncthreads();
    }

    const int gn = wn * 16 + lrow;
    const float bias = b2[gn];
#pragma unroll
    for (int r = 0; r < 8; ++r) {
        const int gm0 = m0 + wm * 32 + r + lhi * 8;
        logits[(size_t)gm0 * DB + gn]        = acc0[r] + bias;
        logits[(size_t)(gm0 + 16) * DB + gn] = acc1[r] + bias;
    }
}

// ---------------- softmax depth + unproject + BEV cell index ----------------
__global__ void depth_idx_kernel(const float* __restrict__ logits,
                                 const float* __restrict__ K,
                                 const float* __restrict__ E,
                                 int* __restrict__ cell_idx)
{
    const int m = blockIdx.x * blockDim.x + threadIdx.x;
    if (m >= HW) return;
    const float* L = logits + (size_t)m * DB;

    float mx = -3.4e38f;
    for (int j = 0; j < DB; ++j) mx = fmaxf(mx, L[j]);
    float s = 0.0f, wsum = 0.0f;
    const float step = 99.0f / 63.0f;
    for (int j = 0; j < DB; ++j) {
        float e = __expf(L[j] - mx);
        s    += e;
        wsum += e * (1.0f + j * step);
    }
    const float d = wsum / s;

    float a = K[0], b = K[1], c = K[2];
    float dd = K[3], e = K[4], f = K[5];
    float g = K[6], h = K[7], i = K[8];
    float det = a * (e * i - f * h) - b * (dd * i - f * g) + c * (dd * h - e * g);
    float inv = 1.0f / det;
    float i00 = (e * i - f * h) * inv, i01 = (c * h - b * i) * inv, i02 = (b * f - c * e) * inv;
    float i10 = (f * g - dd * i) * inv, i11 = (a * i - c * g) * inv, i12 = (c * dd - a * f) * inv;
    float i20 = (dd * h - e * g) * inv, i21 = (b * g - a * h) * inv, i22 = (a * e - b * dd) * inv;

    const float u = (float)(m % IMG_W);
    const float v = (float)(m / IMG_W);
    const float p0 = u * d, p1 = v * d, p2 = d;
    const float cx = i00 * p0 + i01 * p1 + i02 * p2;
    const float cy = i10 * p0 + i11 * p1 + i12 * p2;
    const float cz = i20 * p0 + i21 * p1 + i22 * p2;

    const float wx = E[0] * cx + E[1] * cy + E[2] * cz + E[3];
    const float wy = E[4] * cx + E[5] * cy + E[6] * cz + E[7];

    const int xi = (int)floorf(wx * 2.0f);
    const int yi = (int)floorf((wy + 50.0f) * 2.0f);
    cell_idx[m] = ((unsigned)xi < 200u && (unsigned)yi < 200u) ? (yi * 200 + xi) : -1;
}

// ---------------- splat: scatter-add features into BEV ----------------
__global__ __launch_bounds__(256) void splat_kernel(const float* __restrict__ feat,
                                                    const int* __restrict__ cell_idx,
                                                    float* __restrict__ out)
{
    const int m4 = (blockIdx.x * 256 + threadIdx.x) * 4;   // 4 pixels / thread
    const int c  = blockIdx.y;
    const int4   cells = *(const int4*)&cell_idx[m4];
    const float4 f     = *(const float4*)&feat[(size_t)c * HW + m4];
    float* base = out + (size_t)c * 40000;
    if (cells.x >= 0) unsafeAtomicAdd(base + cells.x, f.x);
    if (cells.y >= 0) unsafeAtomicAdd(base + cells.y, f.y);
    if (cells.z >= 0) unsafeAtomicAdd(base + cells.z, f.z);
    if (cells.w >= 0) unsafeAtomicAdd(base + cells.w, f.w);
}

// ---------------- launcher ----------------
extern "C" void kernel_launch(void* const* d_in, const int* in_sizes, int n_in,
                              void* d_out, int out_size, void* d_ws, size_t ws_size,
                              hipStream_t stream) {
    const float* feat  = (const float*)d_in[0];
    const float* intr  = (const float*)d_in[1];
    const float* extr  = (const float*)d_in[2];
    const float* w1    = (const float*)d_in[3];
    const float* b1    = (const float*)d_in[4];
    const float* gamma = (const float*)d_in[5];
    const float* beta  = (const float*)d_in[6];
    const float* mean  = (const float*)d_in[7];
    const float* var   = (const float*)d_in[8];
    const float* w2    = (const float*)d_in[9];
    const float* b2    = (const float*)d_in[10];
    float* out = (float*)d_out;

    char*  ws  = (char*)d_ws;
    size_t off = 0;
    auto alloc = [&](size_t bytes) {
        size_t o = off;
        off += (bytes + 255) & ~(size_t)255;
        return o;
    };
    unsigned short* featP  = (unsigned short*)(ws + alloc((size_t)HW * C_IN * 2));
    unsigned short* w1P    = (unsigned short*)(ws + alloc((size_t)C_MID * K1 * 2));
    unsigned short* a2     = (unsigned short*)(ws + alloc((size_t)HW * C_MID * 2));
    unsigned short* w2P    = (unsigned short*)(ws + alloc((size_t)DB * C_MID * 2));
    float*          logits = (float*)(ws + alloc((size_t)HW * DB * 4));
    int*            cidx   = (int*)(ws + alloc((size_t)HW * 4));
    unsigned short* zrow   = (unsigned short*)(ws + alloc((size_t)C_IN * 2));

    zero_f32v<<<2048, 256, 0, stream>>>((float4*)out, (size_t)out_size / 4);
    zero_f32v<<<1, 256, 0, stream>>>((float4*)zrow, (size_t)C_IN * 2 / 16);

    cvt_feat_t<<<dim3(HW / 32, C_IN / 32), 256, 0, stream>>>(feat, featP);
    cvt_w1_bf16<<<4096, 256, 0, stream>>>(w1, w1P);
    cvt_w2_bf16<<<(DB * C_MID + 255) / 256, 256, 0, stream>>>(w2, w2P);

    gemm1_conv3x3<<<dim3(HW / BM1, C_MID / BN1), 256, 0, stream>>>(
        featP, w1P, zrow, b1, gamma, beta, mean, var, a2);
    gemm2_conv1x1<<<dim3(HW / 64, 1), 256, 0, stream>>>(a2, w2P, b2, logits);

    depth_idx_kernel<<<HW / 256, 256, 0, stream>>>(logits, intr, extr, cidx);
    splat_kernel<<<dim3(HW / 1024, C_IN), 256, 0, stream>>>(feat, cidx, out);
}
